// MetaMLP_83562883711142
// MI455X (gfx1250) — matrix operand order repaired and checked by reading, not compiled, not run on hardware
//
#include <hip/hip_runtime.h>
#include <hip/hip_bf16.h>
#include <stdint.h>

#define GRU_STEPS 3
#define NN 50000
#define NE 800000
#define DD 128
#define NG 16

typedef __attribute__((ext_vector_type(16))) __bf16 v16bf;
typedef __attribute__((ext_vector_type(8)))  float  v8f;
typedef __attribute__((ext_vector_type(4)))  unsigned int u32x4;
typedef __attribute__((ext_vector_type(8)))  unsigned int u32x8;

union FragBF { v16bf v; u32x4 q[2]; };

__device__ __forceinline__ unsigned short f2bf(float f) {
  unsigned int u = __float_as_uint(f);
  u += 0x7FFFu + ((u >> 16) & 1u);          // round-to-nearest-even
  return (unsigned short)(u >> 16);
}

// ---- Tensor Data Mover: 2D tile (nRows x rowLen8 x 8B) global -> LDS ----
// D# group0 (4 SGPRs) + group1 (8 SGPRs), 2-group form (VADDR2/3 = NULL).
// rowLen8 / rowStride8 in 8-byte units. Compacted into LDS row-after-row.
__device__ __forceinline__ void tdm_load_2d(unsigned ldsAddr, const void* gptr,
                                            unsigned rowLen8, unsigned rowStride8,
                                            unsigned nRows) {
  unsigned long long ga = (unsigned long long)(uintptr_t)gptr;
  u32x4 g0;
  g0[0] = 1u;                                              // count=1, user mode
  g0[1] = ldsAddr;                                         // lds_addr[31:0]
  g0[2] = (unsigned)ga;                                    // global_addr[31:0]
  g0[3] = ((unsigned)(ga >> 32) & 0x01FFFFFFu)             // global_addr[56:32]
          | 0x80000000u;                                   // type=2 ("image")
  u32x8 g1;
  g1[0] = 0x00030000u;                                     // wg_mask=0, data_size=8B
  g1[1] = (rowLen8 & 0xFFFFu) << 16;                       // tensor_dim0[15:0]
  g1[2] = (rowLen8 >> 16) | ((nRows & 0xFFFFu) << 16);     // tensor_dim0[31:16] | tensor_dim1[15:0]
  g1[3] = (nRows >> 16) | ((rowLen8 & 0xFFFFu) << 16);     // tensor_dim1[31:16] | tile_dim0
  g1[4] = (nRows & 0xFFFFu);                               // tile_dim1 | tile_dim2=0
  g1[5] = rowStride8;                                      // tensor_dim0_stride[31:0]
  g1[6] = 0u;                                              // stride0[47:32] | stride1[15:0]
  g1[7] = 0u;
  asm volatile("tensor_load_to_lds %0, %1" : : "s"(g0), "s"(g1) : "memory");
}

__device__ __forceinline__ void wait_tensor0() {
#if __has_builtin(__builtin_amdgcn_s_wait_tensorcnt)
  __builtin_amdgcn_s_wait_tensorcnt(0);
#else
  asm volatile("s_wait_tensorcnt 0x0" ::: "memory");
#endif
}

// ---------------- prep kernels ----------------

__global__ void k_prep_w(const float* __restrict__ We, const float* __restrict__ Wn,
                         const float* __restrict__ Wg,
                         unsigned short* __restrict__ WeT,
                         unsigned short* __restrict__ WnT,
                         unsigned short* __restrict__ WgT) {
  int idx = blockIdx.x * blockDim.x + threadIdx.x;
  const int SZE = 512 * DD, SZN = 384 * DD, SZG = 256 * DD;
  if (idx < SZE) {
    int n = idx & (DD - 1), k = idx >> 7;
    WeT[n * 512 + k] = f2bf(We[k * DD + n]);
  } else if (idx < SZE + SZN) {
    int j = idx - SZE; int n = j & (DD - 1), k = j >> 7;
    WnT[n * 384 + k] = f2bf(Wn[k * DD + n]);
  } else if (idx < SZE + SZN + SZG) {
    int j = idx - SZE - SZN; int n = j & (DD - 1), k = j >> 7;
    WgT[n * 256 + k] = f2bf(Wg[k * DD + n]);
  }
}

__global__ void k_prep_x(const float* __restrict__ x, unsigned short* __restrict__ xb,
                         float* __restrict__ xout) {
  size_t idx = (size_t)blockIdx.x * blockDim.x + threadIdx.x;
  if (idx >= (size_t)NN * DD) return;
  float v = x[idx];
  xb[idx] = f2bf(v);
  size_t i = idx >> 7; int c = (int)(idx & (DD - 1));
  xout[i * 512 + c] = v;
}

__global__ void k_prep_u(const float* __restrict__ u, unsigned short* __restrict__ ub,
                         float* __restrict__ uout) {
  int idx = blockIdx.x * blockDim.x + threadIdx.x;
  if (idx >= NG * DD) return;
  float v = u[idx];
  ub[idx] = f2bf(v);
  int g = idx >> 7, c = idx & (DD - 1);
  uout[g * 512 + c] = v;
}

__global__ void k_prep_ea(const float* __restrict__ ea, unsigned short* __restrict__ eab) {
  size_t idx = (size_t)blockIdx.x * blockDim.x + threadIdx.x;
  if (idx >= (size_t)NE * DD) return;
  eab[idx] = f2bf(ea[idx]);
}

__global__ void k_counts(const int* __restrict__ dst, const int* __restrict__ batch,
                         int* __restrict__ cnt, int* __restrict__ gcnt) {
  int idx = blockIdx.x * blockDim.x + threadIdx.x;
  if (idx < NE) atomicAdd(&cnt[dst[idx]], 1);
  if (idx < NN) atomicAdd(&gcnt[batch[idx]], 1);
}

__global__ void k_aggnorm(const float* __restrict__ agg, const int* __restrict__ cnt,
                          unsigned short* __restrict__ aggb) {
  size_t idx = (size_t)blockIdx.x * blockDim.x + threadIdx.x;
  if (idx >= (size_t)NN * DD) return;
  size_t i = idx >> 7;
  float c = (float)max(cnt[i], 1);
  aggb[idx] = f2bf(agg[idx] / c);
}

// ---------------- edge GEMM: [NE,512] x [512,128], gathered A ----------------
// 256 threads = 8 waves; each wave owns a 16-edge x 128-col tile.
// W^T staged in LDS by the TDM in two 64KB passes (K 0..255, 256..511).
__global__ void __launch_bounds__(256)
k_edge_gemm(const int* __restrict__ src, const int* __restrict__ dst,
            const int* __restrict__ batch,
            const unsigned short* __restrict__ xb,
            unsigned short* __restrict__ eab,          // read old, write new (in place)
            const unsigned short* __restrict__ ub,
            const unsigned short* __restrict__ WeT,    // bf16 [128][512]
            const float* __restrict__ bias,
            float* __restrict__ agg) {
  extern __shared__ unsigned short sW[];               // [128][256] bf16 = 64 KB

  const int wv = threadIdx.x >> 5;
  const int ln = threadIdx.x & 31;
  const int l15 = ln & 15;
  const int hi  = ln >> 4;
  const long long tile = (long long)blockIdx.x * 8 + wv;
  const long long e = tile * 16 + l15;

  const int s  = src[e];
  const int d  = dst[e];
  const int bs = batch[s];

  const unsigned short* pA[4] = {
      xb  + (size_t)s  * DD,
      xb  + (size_t)d  * DD,
      eab + (size_t)e  * DD,
      ub  + (size_t)bs * DD };

  const int c0 = hi ? 8 : 0;   // A layout: lanes 0-15 hold K 0..7 & 16..23

  v8f acc[8];
  v8f zero = {0.f, 0.f, 0.f, 0.f, 0.f, 0.f, 0.f, 0.f};
#pragma unroll
  for (int nt = 0; nt < 8; ++nt) acc[nt] = zero;

#pragma unroll
  for (int p = 0; p < 2; ++p) {                        // fully unrolled: pA[kt>>2]
    if (wv == 0) {                                     // const-folds -> global loads
      // 128 rows x 512B (64 x 8B), global row stride 1024B (128 x 8B)
      tdm_load_2d((unsigned)(uintptr_t)sW, WeT + p * 256, 64u, 128u, 128u);
      wait_tensor0();
    }
    __syncthreads();

#pragma unroll
    for (int kt8 = 0; kt8 < 8; ++kt8) {
      const int kt = p * 8 + kt8;
      const unsigned short* pa = pA[kt >> 2] + ((kt & 3) * 32);
      FragBF a;
      a.q[0] = *(const u32x4*)(pa + c0);
      a.q[1] = *(const u32x4*)(pa + c0 + 16);
#pragma unroll
      for (int nt = 0; nt < 8; ++nt) {
        const unsigned short* pb = sW + ((nt * 16 + l15) * 256 + kt8 * 32 + hi * 8);
        FragBF b;
        b.q[0] = *(const u32x4*)(pb);
        b.q[1] = *(const u32x4*)(pb + 16);
        acc[nt] = __builtin_amdgcn_wmma_f32_16x16x32_bf16(
            false, a.v, false, b.v, (short)0, acc[nt], false, false);
      }
    }
    if (p == 0) __syncthreads();   // protect LDS before restage
  }

  // epilogue: bias + ReLU, bf16 store in place, segment-sum into agg[dst]
  const long long ebase = tile * 16;
#pragma unroll
  for (int nt = 0; nt < 8; ++nt) {
    const int n = nt * 16 + l15;
    const float bz = bias[n];
#pragma unroll
    for (int r = 0; r < 8; ++r) {
      const int m = r + hi * 8;            // C layout: lane row = r + 8*(lane>=16)
      float v = acc[nt][r] + bz;
      v = v > 0.f ? v : 0.f;
      const long long em = ebase + m;
      eab[(size_t)em * DD + n] = f2bf(v);
      const int dm = __shfl(d, m, 32);
      atomicAdd(&agg[(size_t)dm * DD + n], v);
    }
  }
}

// ---------------- node GEMM: [NN,384] x [384,128] ----------------
__global__ void __launch_bounds__(256)
k_node_gemm(const int* __restrict__ batch,
            unsigned short* __restrict__ xb,           // read old x, write new (in place)
            const unsigned short* __restrict__ aggb,
            const unsigned short* __restrict__ ub,
            const unsigned short* __restrict__ WnT,    // bf16 [128][384]
            const float* __restrict__ bias,
            float* __restrict__ xout, int colBase,
            float* __restrict__ gsum) {
  extern __shared__ unsigned short sW[];               // [128][192] bf16 = 48 KB

  const int wv = threadIdx.x >> 5;
  const int ln = threadIdx.x & 31;
  const int l15 = ln & 15;
  const int hi  = ln >> 4;
  const long long tile = (long long)blockIdx.x * 8 + wv;
  const long long nbase = tile * 16;
  const long long row = nbase + l15;
  const long long rc = row < NN ? row : (NN - 1);

  const int bg = batch[rc];

  const unsigned short* pA[3] = {
      xb   + (size_t)rc * DD,
      aggb + (size_t)rc * DD,
      ub   + (size_t)bg * DD };

  const int c0 = hi ? 8 : 0;

  v8f acc[8];
  v8f zero = {0.f, 0.f, 0.f, 0.f, 0.f, 0.f, 0.f, 0.f};
#pragma unroll
  for (int nt = 0; nt < 8; ++nt) acc[nt] = zero;

#pragma unroll
  for (int p = 0; p < 2; ++p) {
    if (wv == 0) {
      // 128 rows x 384B (48 x 8B), global row stride 768B (96 x 8B)
      tdm_load_2d((unsigned)(uintptr_t)sW, WnT + p * 192, 48u, 96u, 128u);
      wait_tensor0();
    }
    __syncthreads();

#pragma unroll
    for (int kt6 = 0; kt6 < 6; ++kt6) {
      const int kt = p * 6 + kt6;
      const unsigned short* pa = pA[kt >> 2] + ((kt & 3) * 32);
      FragBF a;
      a.q[0] = *(const u32x4*)(pa + c0);
      a.q[1] = *(const u32x4*)(pa + c0 + 16);
#pragma unroll
      for (int nt = 0; nt < 8; ++nt) {
        const unsigned short* pb = sW + ((nt * 16 + l15) * 192 + kt6 * 32 + hi * 8);
        FragBF b;
        b.q[0] = *(const u32x4*)(pb);
        b.q[1] = *(const u32x4*)(pb + 16);
        acc[nt] = __builtin_amdgcn_wmma_f32_16x16x32_bf16(
            false, a.v, false, b.v, (short)0, acc[nt], false, false);
      }
    }
    if (p == 0) __syncthreads();
  }

#pragma unroll
  for (int nt = 0; nt < 8; ++nt) {
    const int n = nt * 16 + l15;
    const float bz = bias[n];
#pragma unroll
    for (int r = 0; r < 8; ++r) {
      const int m = r + hi * 8;
      const long long nm = nbase + m;
      if (nm < NN) {
        float v = acc[nt][r] + bz;
        v = v > 0.f ? v : 0.f;
        xout[(size_t)nm * 512 + colBase + n] = v;
        xb[(size_t)nm * DD + n] = f2bf(v);
        const int bm = __shfl(bg, m, 32);
        atomicAdd(&gsum[(size_t)bm * DD + n], v);
      }
    }
  }
}

// ---------------- global GEMM: [16,256] x [256,128] (1 block) ----------------
__global__ void __launch_bounds__(256)
k_glob_gemm(unsigned short* __restrict__ ub,
            const float* __restrict__ gsum, const int* __restrict__ gcnt,
            const unsigned short* __restrict__ WgT,    // bf16 [128][256]
            const float* __restrict__ bias,
            float* __restrict__ uout, int colBase) {
  __shared__ unsigned short sMean[NG * DD];

  for (int i = threadIdx.x; i < NG * DD; i += blockDim.x) {
    int g = i >> 7;
    float c = (float)max(gcnt[g], 1);
    sMean[i] = f2bf(gsum[i] / c);
  }
  __syncthreads();

  if (threadIdx.x >= 32) return;
  const int ln = threadIdx.x;
  const int l15 = ln & 15;
  const int hi  = ln >> 4;
  const int c0 = hi ? 8 : 0;

  v8f acc[8];
  v8f zero = {0.f, 0.f, 0.f, 0.f, 0.f, 0.f, 0.f, 0.f};
#pragma unroll
  for (int nt = 0; nt < 8; ++nt) acc[nt] = zero;

#pragma unroll
  for (int kt = 0; kt < 8; ++kt) {
    const unsigned short* pa = (kt < 4)
        ? (const unsigned short*)(ub + l15 * DD + (kt & 3) * 32)
        : (const unsigned short*)(&sMean[l15 * DD + (kt - 4) * 32]);
    FragBF a;
    a.q[0] = *(const u32x4*)(pa + c0);
    a.q[1] = *(const u32x4*)(pa + c0 + 16);
#pragma unroll
    for (int nt = 0; nt < 8; ++nt) {
      const unsigned short* pb = WgT + ((size_t)(nt * 16 + l15) * 256 + kt * 32 + hi * 8);
      FragBF b;
      b.q[0] = *(const u32x4*)(pb);
      b.q[1] = *(const u32x4*)(pb + 16);
      acc[nt] = __builtin_amdgcn_wmma_f32_16x16x32_bf16(
          false, a.v, false, b.v, (short)0, acc[nt], false, false);
    }
  }

#pragma unroll
  for (int nt = 0; nt < 8; ++nt) {
    const int n = nt * 16 + l15;
    const float bz = bias[n];
#pragma unroll
    for (int r = 0; r < 8; ++r) {
      const int m = r + hi * 8;
      float v = acc[nt][r] + bz;
      v = v > 0.f ? v : 0.f;
      uout[(size_t)m * 512 + colBase + n] = v;
      ub[(size_t)m * DD + n] = f2bf(v);
    }
  }
}

// ---------------- host launch ----------------
extern "C" void kernel_launch(void* const* d_in, const int* in_sizes, int n_in,
                              void* d_out, int out_size, void* d_ws, size_t ws_size,
                              hipStream_t stream) {
  const float* x     = (const float*)d_in[0];
  const int*   eidx  = (const int*)d_in[1];
  const float* ea    = (const float*)d_in[2];
  const float* u     = (const float*)d_in[3];
  const int*   batch = (const int*)d_in[4];
  const float* We    = (const float*)d_in[5];
  const float* be    = (const float*)d_in[6];
  const float* Wn    = (const float*)d_in[7];
  const float* bn    = (const float*)d_in[8];
  const float* Wg    = (const float*)d_in[9];
  const float* bg    = (const float*)d_in[10];

  const int* src = eidx;
  const int* dst = eidx + NE;

  float* out  = (float*)d_out;
  float* uout = out + (size_t)NN * 512;

  char* w = (char*)d_ws;
  size_t off = 0;
  auto carve = [&](size_t bytes) {
    void* p = w + off;
    off = (off + bytes + 255) & ~(size_t)255;
    return p;
  };
  unsigned short* eab  = (unsigned short*)carve((size_t)NE * DD * 2);
  unsigned short* xb   = (unsigned short*)carve((size_t)NN * DD * 2);
  unsigned short* aggb = (unsigned short*)carve((size_t)NN * DD * 2);
  unsigned short* ub   = (unsigned short*)carve((size_t)NG * DD * 2);
  unsigned short* WeT  = (unsigned short*)carve((size_t)DD * 512 * 2);
  unsigned short* WnT  = (unsigned short*)carve((size_t)DD * 384 * 2);
  unsigned short* WgT  = (unsigned short*)carve((size_t)DD * 256 * 2);
  float* agg  = (float*)carve((size_t)NN * DD * 4);
  float* gsum = (float*)carve((size_t)NG * DD * 4);
  int*   cnt  = (int*)carve((size_t)NN * 4);
  int*   gcnt = (int*)carve((size_t)NG * 4);
  (void)ws_size; (void)n_in; (void)in_sizes; (void)out_size;

  hipMemsetAsync(cnt,  0, (size_t)NN * 4, stream);
  hipMemsetAsync(gcnt, 0, (size_t)NG * 4, stream);

  const int T = 256;
  k_prep_w <<<(512*DD + 384*DD + 256*DD + T-1)/T, T, 0, stream>>>(We, Wn, Wg, WeT, WnT, WgT);
  k_prep_x <<<((size_t)NN*DD + T-1)/T, T, 0, stream>>>(x, xb, out);
  k_prep_u <<<(NG*DD + T-1)/T, T, 0, stream>>>(u, ub, uout);
  k_prep_ea<<<((size_t)NE*DD + T-1)/T, T, 0, stream>>>(ea, eab);
  k_counts <<<(NE + T-1)/T, T, 0, stream>>>(dst, batch, cnt, gcnt);

  for (int step = 0; step < GRU_STEPS; ++step) {
    const int colBase = 128 * (step + 1);
    hipMemsetAsync(agg,  0, (size_t)NN * DD * 4, stream);
    hipMemsetAsync(gsum, 0, (size_t)NG * DD * 4, stream);

    k_edge_gemm<<<NE / 128, 256, 128 * 256 * 2, stream>>>(src, dst, batch, xb, eab, ub,
                                                          WeT, be, agg);
    k_aggnorm  <<<((size_t)NN*DD + T-1)/T, T, 0, stream>>>(agg, cnt, aggb);
    k_node_gemm<<<(NN + 127) / 128, 256, 128 * 192 * 2, stream>>>(batch, xb, aggb, ub,
                                                                  WnT, bn, out, colBase, gsum);
    k_glob_gemm<<<1, 256, 0, stream>>>(ub, gsum, gcnt, WgT, bg, uout, colBase);
  }
}
